// PredictYolo_v1_29197187678502
// MI455X (gfx1250) — matrix operand order repaired and checked by reading, not compiled, not run on hardware
//
#include <hip/hip_runtime.h>
#include <hip/hip_bf16.h>
#include <stdint.h>

#define NUM_CLASSES 20
#define CONF_THR    0.05f
#define NMS_THR     0.3f
#define BATCH       32
#define WGRID       13
#define HW          169            // 13*13
#define CH          25
#define NTOT        (BATCH * HW)   // 5408
#define NPAD        8192           // next pow2 >= NTOT
#define NEG_INF_F   (-3.402823466e38f)

typedef __attribute__((ext_vector_type(16))) _Float16     v16h;
typedef __attribute__((ext_vector_type(8)))  float        v8f;
typedef __attribute__((ext_vector_type(4)))  unsigned int u32x4;
typedef __attribute__((ext_vector_type(4)))  int          i32x4;
typedef __attribute__((ext_vector_type(8)))  int          i32x8;

#ifndef __has_builtin
#define __has_builtin(x) 0
#endif
#if __has_builtin(__builtin_amdgcn_tensor_load_to_lds)
#define HAS_TDM 1
#else
#define HAS_TDM 0
#endif

#ifndef __has_include
#define __has_include(x) 0
#endif
#if __has_include(<hip/amd_detail/amd_gfx1250_TDM.h>)
#define TDM_6ARG 1
#else
#define TDM_6ARG 0
#endif

__device__ __forceinline__ float sigmoidf_(float x) {
    return 1.0f / (1.0f + __expf(-x));
}

#if HAS_TDM
// Issue one 1-D TDM transfer: ndw dwords from global -> LDS. Call from one wave.
__device__ __forceinline__ void tdm_load_dwords_to_lds(void* ldsDst, const void* gsrc,
                                                       unsigned int ndw) {
    unsigned long long ga = (unsigned long long)(uintptr_t)gsrc;
    unsigned int laddr = (unsigned int)(uintptr_t)ldsDst;  // low 32 bits = LDS byte offset
    u32x4 g0;
    g0.x = 1u;                                             // count=1 valid, is_restore=0
    g0.y = laddr;                                          // lds_addr
    g0.z = (unsigned int)(ga & 0xffffffffull);             // global_addr[31:0]
    g0.w = (unsigned int)((ga >> 32) & 0x1ffffffull)       // global_addr[56:32]
         | (2u << 30);                                     // type = 2 ("image")
    i32x8 g1;
    g1[0] = (int)(2u << 16);                               // data_size = 4 bytes
    g1[1] = (int)((ndw & 0xffffu) << 16);                  // tensor_dim0[15:0] @ bits[63:48]
    g1[2] = (int)(((ndw >> 16) & 0xffffu) | (1u << 16));   // tensor_dim0[31:16] | tensor_dim1=1
    g1[3] = (int)((ndw & 0xffffu) << 16);                  // tile_dim0 = ndw (<= 65535)
    g1[4] = 1;                                             // tile_dim1 = 1, tile_dim2 = 0
    g1[5] = (int)ndw;                                      // tensor_dim0_stride[31:0]
    g1[6] = 0;
    g1[7] = 0;
    i32x4 gz = (i32x4)0;
#if TDM_6ARG
    i32x8 gz8 = (i32x8)0;
    __builtin_amdgcn_tensor_load_to_lds(g0, g1, gz, gz, gz8, 0);
#else
    __builtin_amdgcn_tensor_load_to_lds(g0, g1, gz, gz, 0);
#endif
}
#endif

// ---------------------------------------------------------------------------
// Kernel 1: decode. One block per batch image. 11 waves; wave w owns cells
// [16w, 16w+16). WMMA computes the 20-class softmax denominator (row sums of
// exp(logit - max)) for 16 cells per wave in one v_wmma_f32_16x16x32_f16.
// ---------------------------------------------------------------------------
__global__ __launch_bounds__(352) void yolo_decode_wmma(const float* __restrict__ pyolos,
                                                        float* __restrict__ out,
                                                        float* __restrict__ ws) {
    __shared__ float sp[CH * HW];    // 25 x 169 slice
    __shared__ float smax[11 * 16];  // per-cell max logit
    __shared__ float sden[11 * 16];  // per-cell softmax denominator

    const int tid  = threadIdx.x;
    const int b    = blockIdx.x;
    const int wave = tid >> 5;
    const int lane = tid & 31;

    const float* src = pyolos + (size_t)b * (CH * HW);

#if HAS_TDM
    if (tid < 32) {
        tdm_load_dwords_to_lds(sp, src, CH * HW);
        __builtin_amdgcn_s_wait_tensorcnt(0);
    }
#else
    for (int n = tid; n < CH * HW; n += 352) sp[n] = src[n];
#endif
    __syncthreads();

    float* out_ids    = out;
    float* out_boxes  = out + NTOT;
    float* out_labels = out + 5 * NTOT;
    float* out_scores = out + 6 * NTOT;
    float* oboxes     = ws;               // group-offset boxes for NMS
    float* skeyG      = ws + 4 * NTOT;    // sort keys

    const int  r      = lane & 15;
    const int  cell   = wave * 16 + r;
    const bool cvalid = (cell < HW);

    float pconf = 0.0f, m = NEG_INF_F;
    int amax = 0;
    if (lane < 16) {
        if (cvalid) {
            pconf = sigmoidf_(sp[cell]);  // channel 0 = objectness logit
            for (int ch = 0; ch < NUM_CLASSES; ++ch) {
                float l = sp[(1 + ch) * HW + cell];
                if (l > m) { m = l; amax = ch; }
            }
            smax[cell] = m;
        } else {
            smax[wave * 16 + r] = 0.0f;
        }
    }
    __syncthreads();

    // Build A = exp(logit - max) in the 16x32 f16 A-layout; B = ones for K<20.
    const float mm  = smax[wave * 16 + r];
    const bool  low = (lane < 16);
    v16h a;
#pragma unroll
    for (int e = 0; e < 16; ++e) {
        int K = low ? (e < 8 ? e : e + 8) : (e < 8 ? e + 8 : e + 16);
        float v = 0.0f;
        if (cvalid && K < NUM_CLASSES) v = __expf(sp[(1 + K) * HW + cell] - mm);
        a[e] = (_Float16)v;
    }
    v16h bb;
#pragma unroll
    for (int e = 0; e < 16; ++e) {
        bool one = low ? true : (e < 8);  // lane>=16 covers K=8..15 in first 8 halves
        bb[e] = one ? (_Float16)1.0f : (_Float16)0.0f;
    }
    v8f cacc = {};
    v8f d = __builtin_amdgcn_wmma_f32_16x16x32_f16(false, a, false, bb,
                                                   (short)0, cacc, false, false);
    if (lane == 0) {
#pragma unroll
        for (int g = 0; g < 8; ++g) sden[wave * 16 + g] = d[g];       // rows 0..7
    } else if (lane == 16) {
#pragma unroll
        for (int g = 0; g < 8; ++g) sden[wave * 16 + 8 + g] = d[g];   // rows 8..15
    }
    __syncthreads();

    if (lane < 16 && cvalid) {
        float denom    = sden[cell];
        float cls_conf = 1.0f / denom;         // exp(max-max)/sum = 1/denom
        float score    = cls_conf * pconf;
        int   label    = amax + 1;

        float tx = sp[(1 + NUM_CLASSES + 0) * HW + cell];
        float ty = sp[(1 + NUM_CLASSES + 1) * HW + cell];
        float tw = sp[(1 + NUM_CLASSES + 2) * HW + cell];
        float th = sp[(1 + NUM_CLASSES + 3) * HW + cell];
        float gx = (float)(cell % WGRID);
        float gy = (float)(cell / WGRID);
        float cx = (sigmoidf_(tx) + gx) / (float)WGRID;
        float cy = (sigmoidf_(ty) + gy) / (float)WGRID;
        float bw = sigmoidf_(tw);
        float bh = sigmoidf_(th);
        float x0 = fminf(fmaxf(cx - bw * 0.5f, 0.0f), 1.0f);
        float y0 = fminf(fmaxf(cy - bh * 0.5f, 0.0f), 1.0f);
        float x1 = fminf(fmaxf(cx + bw * 0.5f, 0.0f), 1.0f);
        float y1 = fminf(fmaxf(cy + bh * 0.5f, 0.0f), 1.0f);

        int n = b * HW + cell;
        out_ids[n]    = (float)b;
        out_labels[n] = (float)label;
        out_scores[n] = score;
        out_boxes[n * 4 + 0] = x0;
        out_boxes[n * 4 + 1] = y0;
        out_boxes[n * 4 + 2] = x1;
        out_boxes[n * 4 + 3] = y1;

        float grp = (float)((b * (NUM_CLASSES + 2) + label) * 4);  // group offset
        oboxes[n * 4 + 0] = x0 + grp;
        oboxes[n * 4 + 1] = y0 + grp;
        oboxes[n * 4 + 2] = x1 + grp;
        oboxes[n * 4 + 3] = y1 + grp;

        bool valid = score > CONF_THR;
        skeyG[n] = valid ? score : NEG_INF_F;
    }
}

// ---------------------------------------------------------------------------
// Kernel 2: single-workgroup bitonic sort of (key desc, idx asc) over 8192
// padded slots (64 KB LDS). Gathers offset-boxes into sorted order for NMS.
// ---------------------------------------------------------------------------
__global__ __launch_bounds__(1024) void yolo_sort(float* __restrict__ ws) {
    __shared__ float skey[NPAD];
    __shared__ int   sidx[NPAD];
    const int tid = threadIdx.x;

    float*        oboxes = ws;
    float*        skeyG  = ws + 4 * NTOT;
    unsigned int* orderG = (unsigned int*)(ws + 5 * NTOT);
    unsigned int* svalG  = orderG + NTOT;
    float*        sboxG  = (float*)(svalG + NTOT);

#if HAS_TDM
    if (tid < 32) {
        tdm_load_dwords_to_lds(skey, skeyG, NTOT);
        __builtin_amdgcn_s_wait_tensorcnt(0);
    }
    for (int n = NTOT + tid; n < NPAD; n += 1024) skey[n] = NEG_INF_F;
#else
    for (int n = tid; n < NPAD; n += 1024) skey[n] = (n < NTOT) ? skeyG[n] : NEG_INF_F;
#endif
    for (int n = tid; n < NPAD; n += 1024) sidx[n] = n;
    __syncthreads();

    for (int k = 2; k <= NPAD; k <<= 1) {
        for (int j = k >> 1; j > 0; j >>= 1) {
#pragma unroll
            for (int s = 0; s < NPAD / 1024; ++s) {
                int i = tid + s * 1024;
                int l = i ^ j;
                if (l > i) {
                    float ki = skey[i], kl = skey[l];
                    int   ii = sidx[i], il = sidx[l];
                    // iFirst: element at i belongs earlier in final order
                    bool iFirst = (ki > kl) || ((ki == kl) && (ii < il));
                    bool up     = ((i & k) == 0);
                    if (up != iFirst) {
                        skey[i] = kl; skey[l] = ki;
                        sidx[i] = il; sidx[l] = ii;
                    }
                }
            }
            __syncthreads();
        }
    }

    for (int n = tid; n < NTOT; n += 1024) {
        int o = sidx[n];
        orderG[n] = (unsigned int)o;
        svalG[n]  = (skey[n] > -1.0e30f) ? 1u : 0u;
        sboxG[n * 4 + 0] = oboxes[o * 4 + 0];
        sboxG[n * 4 + 1] = oboxes[o * 4 + 1];
        sboxG[n * 4 + 2] = oboxes[o * 4 + 2];
        sboxG[n * 4 + 3] = oboxes[o * 4 + 3];
    }
}

// ---------------------------------------------------------------------------
// Kernel 3: single-workgroup greedy NMS, fully LDS-resident (sorted boxes
// 86.5 KB + flags). TDM stages boxes; per step the whole block suppresses.
// ---------------------------------------------------------------------------
__global__ __launch_bounds__(1024) void yolo_nms(const float* __restrict__ ws,
                                                 float* __restrict__ out) {
    __shared__ float         nb[NTOT * 4];
    __shared__ unsigned char kf[NTOT];
    const int tid = threadIdx.x;

    const unsigned int* orderG = (const unsigned int*)(ws + 5 * NTOT);
    const unsigned int* svalG  = orderG + NTOT;
    const float*        sboxG  = (const float*)(svalG + NTOT);
    float*              out_keep = out + 7 * NTOT;

#if HAS_TDM
    if (tid < 32) {
        tdm_load_dwords_to_lds(nb, sboxG, NTOT * 4);
        __builtin_amdgcn_s_wait_tensorcnt(0);
    }
#else
    for (int n = tid; n < NTOT * 4; n += 1024) nb[n] = sboxG[n];
#endif
    for (int n = tid; n < NTOT; n += 1024) kf[n] = (unsigned char)svalG[n];
    __syncthreads();

    for (int i = 0; i < NTOT; ++i) {
        if (kf[i]) {  // uniform across block (same LDS word after barrier)
            float x0 = nb[i * 4 + 0], y0 = nb[i * 4 + 1];
            float x1 = nb[i * 4 + 2], y1 = nb[i * 4 + 3];
            float ai = (x1 - x0) * (y1 - y0);
            for (int j = i + 1 + tid; j < NTOT; j += 1024) {
                if (!kf[j]) continue;
                float jx0 = nb[j * 4 + 0], jy0 = nb[j * 4 + 1];
                float jx1 = nb[j * 4 + 2], jy1 = nb[j * 4 + 3];
                float u0 = fmaxf(x0, jx0), v0 = fmaxf(y0, jy0);
                float u1 = fminf(x1, jx1), v1 = fminf(y1, jy1);
                float w = fmaxf(u1 - u0, 0.0f);
                float h = fmaxf(v1 - v0, 0.0f);
                float inter = w * h;
                float aj = (jx1 - jx0) * (jy1 - jy0);
                float iou = inter / (ai + aj - inter + 1e-9f);
                if (iou > NMS_THR) kf[j] = 0;
            }
        }
        __syncthreads();
    }

    // Scatter back through inverse permutation: keep[order[n]] = kf_sorted[n].
    for (int n = tid; n < NTOT; n += 1024) {
        out_keep[orderG[n]] = kf[n] ? 1.0f : 0.0f;
    }
}

extern "C" void kernel_launch(void* const* d_in, const int* in_sizes, int n_in,
                              void* d_out, int out_size, void* d_ws, size_t ws_size,
                              hipStream_t stream) {
    (void)in_sizes; (void)n_in; (void)out_size; (void)ws_size;
    const float* pyolos = (const float*)d_in[0];
    float* out = (float*)d_out;
    float* ws  = (float*)d_ws;

    yolo_decode_wmma<<<dim3(BATCH), dim3(352), 0, stream>>>(pyolos, out, ws);
    yolo_sort<<<dim3(1), dim3(1024), 0, stream>>>(ws);
    yolo_nms<<<dim3(1), dim3(1024), 0, stream>>>(ws, out);
}
